// ThreeDElasticityNet_Cheb_71064528880234
// MI455X (gfx1250) — matrix operand order repaired and checked by reading, compiled, not run on hardware
//
#include <hip/hip_runtime.h>
#include <hip/hip_bf16.h>

typedef __attribute__((ext_vector_type(16))) __bf16    v16bf;
typedef __attribute__((ext_vector_type(8)))  float     v8f;
typedef __attribute__((ext_vector_type(8)))  unsigned  v8u;

#define N1V    30000
#define NEDGE  240000
#define KCHEB  10
#define MAXC   256

// ---------------- elementwise / graph kernels ----------------

__global__ void k_fill0(float* p, int n) {
    int i = blockIdx.x * blockDim.x + threadIdx.x;
    if (i < n) p[i] = 0.0f;
}

__global__ void k_negcopy(float* __restrict__ d, const float* __restrict__ s, int n) {
    int i = blockIdx.x * blockDim.x + threadIdx.x;
    if (i < n) d[i] = -s[i];
}

__global__ void k_relu_copy(float* __restrict__ d, const float* __restrict__ s, int n) {
    int i = blockIdx.x * blockDim.x + threadIdx.x;
    if (i < n) d[i] = fmaxf(s[i], 0.0f);
}

__global__ void k_bias_init(float* __restrict__ out, const float* __restrict__ bias,
                            int co, int n) {
    int i = blockIdx.x * blockDim.x + threadIdx.x;
    if (i < n) out[i] = bias[i % co];
}

// layer-0 input, feature dim padded 3 -> 32 with zeros
__global__ void k_extract_x(float* __restrict__ h, const float* __restrict__ x, int b) {
    int i = blockIdx.x * blockDim.x + threadIdx.x;   // over N1*32
    if (i < N1V * 32) {
        int n = i >> 5, c = i & 31;
        h[i] = (c < 3) ? x[n * 9 + b * 3 + c] : 0.0f;
    }
}

__global__ void k_deg(const int* __restrict__ src, const int* __restrict__ dst,
                      float* __restrict__ deg) {
    int e = blockIdx.x * blockDim.x + threadIdx.x;
    if (e < NEDGE) {
        int s = src[e];
        if (s != dst[e]) atomicAdd(&deg[s], 1.0f);
    }
}

__global__ void k_dinv(const float* __restrict__ deg, float* __restrict__ dinv) {
    int n = blockIdx.x * blockDim.x + threadIdx.x;
    if (n < N1V) {
        float d = deg[n];
        dinv[n] = (d > 0.0f) ? rsqrtf(fmaxf(d, 1.0f)) : 0.0f;
    }
}

__global__ void k_wedge(const int* __restrict__ src, const int* __restrict__ dst,
                        const float* __restrict__ dinv, float* __restrict__ w) {
    int e = blockIdx.x * blockDim.x + threadIdx.x;
    if (e < NEDGE) {
        int s = src[e], d = dst[e];
        w[e] = (s != d) ? (-dinv[s] * dinv[d]) : 0.0f;
    }
}

// Tout[dst, c] += scale * w_edge[e] * Tin[src, c]  (edge x feature parallel)
__global__ void k_spmm(const int* __restrict__ src, const int* __restrict__ dst,
                       const float* __restrict__ w,
                       const float* __restrict__ Tin, float* __restrict__ Tout,
                       int ci, float scale) {
    int i = blockIdx.x * blockDim.x + threadIdx.x;
    int tot = NEDGE * ci;
    if (i < tot) {
        int e = i / ci, c = i % ci;
        float wv = w[e] * scale;
        if (wv != 0.0f)
            atomicAdd(&Tout[dst[e] * ci + c], wv * Tin[src[e] * ci + c]);
    }
}

__global__ void k_final(const float* __restrict__ out, float* __restrict__ y, int b) {
    int n = blockIdx.x * blockDim.x + threadIdx.x;
    if (n < N1V) y[n * 3 + b] = out[n];    // co == 1 on last layer
}

// ---------------- weight packing: fp32 (10,ci,co) -> bf16 fragment layout ----------
// Packed dword index: (((kc*ktiles + t)*coPad + col)*2 + half)*8 + v
//   holds bf16x2 = W[kb, col], W[kb+1, col] with kb = t*32 + half*16 + 2v
// Zero-padded for kb >= ci (layer 0) and col >= co (last layer).

__global__ void k_pack_w(const float* __restrict__ W, unsigned* __restrict__ out,
                         int ci, int co, int ciPad, int coPad, int total) {
    int i = blockIdx.x * blockDim.x + threadIdx.x;
    if (i >= total) return;
    int v    = i & 7;
    int half = (i >> 3) & 1;
    int col  = (i >> 4) % coPad;
    int r    = (i >> 4) / coPad;
    int ktiles = ciPad >> 5;
    int t  = r % ktiles;
    int kc = r / ktiles;
    int kb = (t << 5) + (half << 4) + (v << 1);
    float b0 = (kb     < ci && col < co) ? W[(kc * ci + kb    ) * co + col] : 0.0f;
    float b1 = (kb + 1 < ci && col < co) ? W[(kc * ci + kb + 1) * co + col] : 0.0f;
    unsigned short u0 = __builtin_bit_cast(unsigned short, (__bf16)b0);
    unsigned short u1 = __builtin_bit_cast(unsigned short, (__bf16)b1);
    out[i] = ((unsigned)u1 << 16) | (unsigned)u0;
}

// ---------------- WMMA GEMM: C[n1 x co] += A[n1 x ciPad] @ Bpacked ----------------
// One wave per 16x16 C tile. Unguarded float4 A loads (fp32 -> bf16 in-register),
// unguarded 32B packed-bf16 B loads, f32 accumulation in v_wmma_f32_16x16x32_bf16.

__global__ __launch_bounds__(256) void k_gemm_wmma(
    const float* __restrict__ A, const unsigned* __restrict__ Bp, float* __restrict__ C,
    int ciPad, int co, int colTiles, int totalTiles, int coPad)
{
    int lane = threadIdx.x & 31;
    int wid  = threadIdx.x >> 5;
    int tile = blockIdx.x * 8 + wid;           // wave-uniform
    if (tile >= totalTiles) return;

    int rowT = tile / colTiles;
    int colT = tile - rowT * colTiles;
    int row0 = rowT << 4;
    int col0 = colT << 4;

    int half = lane >> 4;
    int nl   = lane & 15;
    int arow = row0 + nl;        // A row this lane supplies
    int bcol = col0 + nl;        // B/C column this lane touches
    bool cOK = (bcol < co);      // only false on last layer (co == 1)

    // C/D layout: VGPR r holds M = r + 8*half, N = nl
    v8f c;
#pragma unroll
    for (int r = 0; r < 8; ++r) {
        int m = row0 + r + half * 8;
        c[r] = cOK ? C[m * co + bcol] : 0.0f;
    }

    const float4* Arow = (const float4*)(A + arow * ciPad);

    for (int kk = 0; kk < ciPad; kk += 32) {
        // A: K = [kk+half*16, +16) -> 4x b128
        int f4 = (kk >> 2) + (half << 2);
        float4 a0 = Arow[f4];
        float4 a1 = Arow[f4 + 1];
        float4 a2 = Arow[f4 + 2];
        float4 a3 = Arow[f4 + 3];
        v16bf a;
        a[0]  = (__bf16)a0.x; a[1]  = (__bf16)a0.y; a[2]  = (__bf16)a0.z; a[3]  = (__bf16)a0.w;
        a[4]  = (__bf16)a1.x; a[5]  = (__bf16)a1.y; a[6]  = (__bf16)a1.z; a[7]  = (__bf16)a1.w;
        a[8]  = (__bf16)a2.x; a[9]  = (__bf16)a2.y; a[10] = (__bf16)a2.z; a[11] = (__bf16)a2.w;
        a[12] = (__bf16)a3.x; a[13] = (__bf16)a3.y; a[14] = (__bf16)a3.z; a[15] = (__bf16)a3.w;

        // B: 8 contiguous packed dwords -> 2x b128, already bf16
        const v8u* bp = (const v8u*)(Bp + (((((kk >> 5) * coPad + bcol) << 1) + half) << 3));
        v16bf b = __builtin_bit_cast(v16bf, *bp);

        c = __builtin_amdgcn_wmma_f32_16x16x32_bf16(
                /*neg_a=*/false, a, /*neg_b=*/false, b,
                /*c_mod=*/(short)0, c, /*reuse_a=*/false, /*reuse_b=*/false);
    }

#pragma unroll
    for (int r = 0; r < 8; ++r) {
        int m = row0 + r + half * 8;
        if (cOK) C[m * co + bcol] = c[r];
    }
}

// ---------------- host-side orchestration ----------------

static inline dim3 g1(long long n) { return dim3((unsigned)((n + 255) / 256)); }

extern "C" void kernel_launch(void* const* d_in, const int* in_sizes, int n_in,
                              void* d_out, int out_size, void* d_ws, size_t ws_size,
                              hipStream_t stream) {
    const float* x   = (const float*)d_in[0];
    const int*   ei  = (const int*)d_in[1];
    const int*   src = ei;
    const int*   dsi = ei + NEDGE;

    static const int DI [8] = {3, 32, 64, 128, 256, 128, 64, 32};   // actual ci
    static const int DIP[8] = {32, 32, 64, 128, 256, 128, 64, 32};  // padded ci
    static const int DO [8] = {32, 64, 128, 256, 128, 64, 32, 1};   // actual co
    static const int DOP[8] = {32, 64, 128, 256, 128, 64, 32, 16};  // padded co

    // workspace carve-up (~130 MB)
    char* w = (char*)d_ws;
    size_t tb = (size_t)N1V * MAXC * sizeof(float);
    float* TB0 = (float*)w; w += tb;
    float* TB1 = (float*)w; w += tb;
    float* TB2 = (float*)w; w += tb;
    float* OUT = (float*)w; w += tb;
    float* deg  = (float*)w; w += (size_t)N1V * sizeof(float);
    float* dinv = (float*)w; w += (size_t)N1V * sizeof(float);
    float* wed  = (float*)w; w += (size_t)NEDGE * sizeof(float);
    w = (char*)(((uintptr_t)w + 255) & ~(uintptr_t)255);
    unsigned* BP = (unsigned*)w;                       // packed bf16 weights

    // pack all 24 weight tensors (bf16, fragment order, zero-padded)
    size_t wOff[24];
    {
        size_t off = 0;
        for (int b = 0; b < 3; ++b)
            for (int l = 0; l < 8; ++l) {
                int idx = b * 8 + l;
                wOff[idx] = off;
                int total = KCHEB * (DIP[l] >> 5) * DOP[l] * 16;   // dwords
                k_pack_w<<<g1(total), 256, 0, stream>>>(
                    (const float*)d_in[2 + idx], BP + off,
                    DI[l], DO[l], DIP[l], DOP[l], total);
                off += (size_t)total;
            }
    }

    // normalized edge weights: w_e = -dinv[src]*dinv[dst]*(src!=dst)
    k_fill0<<<g1(N1V), 256, 0, stream>>>(deg, N1V);
    k_deg  <<<g1(NEDGE), 256, 0, stream>>>(src, dsi, deg);
    k_dinv <<<g1(N1V), 256, 0, stream>>>(deg, dinv);
    k_wedge<<<g1(NEDGE), 256, 0, stream>>>(src, dsi, dinv, wed);

    for (int b = 0; b < 3; ++b) {
        float *t0 = TB0, *t1 = TB1, *t2 = TB2;
        k_extract_x<<<g1((long long)N1V * 32), 256, 0, stream>>>(t0, x, b);

        for (int l = 0; l < 8; ++l) {
            int ci = DIP[l], co = DO[l], coPad = DOP[l];
            const unsigned* Wp   = BP + wOff[b * 8 + l];
            const float*    bias = (const float*)d_in[2 + 24 + b * 8 + l];
            int nco = N1V * co;
            int nci = N1V * ci;
            int perK = (ci >> 5) * coPad * 16;         // packed dwords per Chebyshev k

            int colTiles   = coPad >> 4;
            int totalTiles = (N1V >> 4) * colTiles;
            dim3 gg((unsigned)((totalTiles + 7) / 8));

            // out = bias ; k = 0 : out += T0 @ W[0]
            k_bias_init<<<g1(nco), 256, 0, stream>>>(OUT, bias, co, nco);
            k_gemm_wmma<<<gg, 256, 0, stream>>>(t0, Wp, OUT, ci, co, colTiles, totalTiles, coPad);

            // k = 1 : T1 = prop(T0); out += T1 @ W[1]
            k_fill0<<<g1(nci), 256, 0, stream>>>(t1, nci);
            k_spmm <<<g1((long long)NEDGE * ci), 256, 0, stream>>>(src, dsi, wed, t0, t1, ci, 1.0f);
            k_gemm_wmma<<<gg, 256, 0, stream>>>(t1, Wp + perK, OUT, ci, co, colTiles, totalTiles, coPad);

            // k = 2..9 : T2 = 2*prop(T1) - T0; out += T2 @ W[k]; rotate
            for (int k = 2; k < KCHEB; ++k) {
                k_negcopy<<<g1(nci), 256, 0, stream>>>(t2, t0, nci);
                k_spmm   <<<g1((long long)NEDGE * ci), 256, 0, stream>>>(src, dsi, wed, t1, t2, ci, 2.0f);
                k_gemm_wmma<<<gg, 256, 0, stream>>>(t2, Wp + (size_t)k * perK, OUT,
                                                    ci, co, colTiles, totalTiles, coPad);
                float* tmp = t0; t0 = t1; t1 = t2; t2 = tmp;
            }

            if (l < 7) {
                k_relu_copy<<<g1(nco), 256, 0, stream>>>(t0, OUT, nco);   // next layer's h
            } else {
                k_final<<<g1(N1V), 256, 0, stream>>>(OUT, (float*)d_out, b);
            }
        }
    }
    (void)in_sizes; (void)n_in; (void)out_size; (void)ws_size;
}
